// FNO_40450001993791
// MI455X (gfx1250) — matrix operand order repaired and checked by reading, not compiled, not run on hardware
//
#include <hip/hip_runtime.h>
#include <math.h>

// ---------------------------------------------------------------------------
// FNO (spectral conv stack) for MI455X / gfx1250, wave32 + WMMA f16->f32.
// ---------------------------------------------------------------------------

typedef __attribute__((ext_vector_type(16))) _Float16 v16h;
typedef __attribute__((ext_vector_type(8)))  float    v8f;

#define NPTS   32768          // 32^3
#define BB     2
#define CC     128            // model channels
#define FF     128            // frequencies
#define DD     3
#define KSPLIT 32             // split-K slices for forward GEMM
#define KCHUNK (NPTS / KSPLIT) // 1024
#define NTPB   8              // n-tiles per block in the inverse GEMM
#define TWO_PI_F 6.28318530717958647692f

// ---- workspace layout (bytes) ---------------------------------------------
#define OFF_HF32  ((size_t)0)                         // B*N*C f32   = 33.55 MB
#define OFF_HF16  (OFF_HF32 + (size_t)BB*NPTS*CC*4)   // B*N*C f16   = 16.78 MB
#define OFF_COS   (OFF_HF16 + (size_t)BB*NPTS*CC*2)   // B*N*F f16   = 16.78 MB
#define OFF_SIN   (OFF_COS  + (size_t)BB*NPTS*FF*2)   // B*N*F f16   = 16.78 MB
#define OFF_HPART (OFF_SIN  + (size_t)BB*NPTS*FF*2)   // KS*B*2*F*C f32 = 8.39 MB
#define OFF_HTR   (OFF_HPART + (size_t)KSPLIT*BB*2*FF*CC*4)
#define OFF_HTI   (OFF_HTR  + (size_t)BB*FF*CC*4)
#define OFF_GTR   (OFF_HTI  + (size_t)BB*FF*CC*4)
#define OFF_GTI   (OFF_GTR  + (size_t)BB*FF*CC*2)

// ---------------------------------------------------------------------------
// WMMA per-lane gathers (wave32 layouts per CDNA5 ISA 7.12.2).
// LDS tiles are stored transposed: t[row=M or N'][col=K], K contiguous so A
// K-pairs are contiguous 32-bit LDS reads. Pad to 34 to spread banks.
// ---------------------------------------------------------------------------
__device__ __forceinline__ v16h gatherA_lds(const _Float16 (*t)[34], int lane) {
  int m  = lane & 15;
  int kb = (lane >> 4) << 3;    // lanes 16-31 hold K+8
  v16h a;
#pragma unroll
  for (int j = 0; j < 8; ++j) { // VGPR j: K = grp*16 + kb + (j&3)*2 (+1)
    int k = ((j >> 2) << 4) + kb + ((j & 3) << 1);
    a[2 * j]     = t[m][k];
    a[2 * j + 1] = t[m][k + 1];
  }
  return a;
}

__device__ __forceinline__ v16h gatherB_lds(const _Float16 (*t)[34], int lane) {
  int np = lane & 15;
  int kb = (lane >> 4) << 3;    // lanes 16-31 hold K+8 (same order as A)
  v16h b;
#pragma unroll
  for (int i = 0; i < 16; ++i) b[i] = t[np][kb + i + ((i >> 3) << 3)];
  return b;
}

__device__ __forceinline__ float gelu_tanh(float x) {
  float x3 = x * x * x;
  float t  = tanhf(0.7978845608028654f * (x + 0.044715f * x3));
  return 0.5f * x * (1.0f + t);
}

// ---------------------------------------------------------------------------
// 1) fc_in: hf = a @ W_in + b  -> f32 (state) and f16 (WMMA operand copy)
// ---------------------------------------------------------------------------
__global__ void fno_fc_in(const float* __restrict__ a, const float* __restrict__ w,
                          const float* __restrict__ bias,
                          float* __restrict__ hf32, _Float16* __restrict__ hf16) {
  int tid = blockIdx.x * blockDim.x + threadIdx.x;   // over B*N*C
  if (tid >= BB * NPTS * CC) return;
  int c   = tid & (CC - 1);
  int row = tid >> 7;                                 // b*N + n
  const float* ar = a + (size_t)row * 4;
  float acc = bias[c];
#pragma unroll
  for (int i = 0; i < 4; ++i) acc += ar[i] * w[i * CC + c];
  hf32[tid] = acc;
  hf16[tid] = (_Float16)acc;
}

// ---------------------------------------------------------------------------
// 2) phase tables: ang = 2*pi*(pts . k_f);  cos/sin -> f16 (B,N,F) row-major
// ---------------------------------------------------------------------------
__global__ void fno_phi(const float* __restrict__ pts, const float* __restrict__ k,
                        _Float16* __restrict__ c16, _Float16* __restrict__ s16) {
  int tid = blockIdx.x * blockDim.x + threadIdx.x;   // over B*N*F
  if (tid >= BB * NPTS * FF) return;
  int f   = tid & (FF - 1);
  int row = tid >> 7;
  const float* p  = pts + (size_t)row * DD;
  const float* kf = k + (size_t)f * DD;
  float ang = TWO_PI_F * (p[0] * kf[0] + p[1] * kf[1] + p[2] * kf[2]);
  float s, c;
  __sincosf(ang, &s, &c);
  c16[tid] = (_Float16)c;
  s16[tid] = (_Float16)s;
}

// ---------------------------------------------------------------------------
// 3) forward transform (split-K, deterministic): per slice ks,
//    Hpart[ks,b,{r,i},f,c] = sum_{n in slice} cos/sin[n,f] * hf16[n,c]
//    M=f tile, N'=c tile, K=n. Block = 4 waves sharing cos/sin A-tiles;
//    each wave owns TWO c-tiles -> 4 WMMAs per A-stage.
// ---------------------------------------------------------------------------
__global__ void __launch_bounds__(128) fno_fwd(
    const _Float16* __restrict__ hf16, const _Float16* __restrict__ c16,
    const _Float16* __restrict__ s16, float* __restrict__ hpart) {
  __shared__ _Float16 At[2][16][34];  // [cos/sin][m=f][k=n]
  __shared__ _Float16 Bt[8][16][34];  // per c-tile [n'=c][k=n]
  int lane = threadIdx.x & 31;
  int wave = threadIdx.x >> 5;
  int ks   = blockIdx.x;              // 0..KSPLIT-1
  int f0   = blockIdx.y * 16;         // f tile 0..7
  int b    = blockIdx.z;
  int c0   = wave * 32;               // two 16-wide c-tiles per wave

  const _Float16* cosB = c16 + (size_t)b * NPTS * FF + f0;
  const _Float16* sinB = s16 + (size_t)b * NPTS * FF + f0;
  const _Float16* hfB  = hf16 + (size_t)b * NPTS * CC + c0;
  int k0 = ks * KCHUNK;

  v8f aR0 = {}, aR1 = {}, aI0 = {}, aI1 = {};
  for (int kk = 0; kk < KCHUNK; kk += 32) {
    size_t krow = (size_t)(k0 + kk + lane);
    if (wave == 0) {
      const _Float16* src = cosB + krow * FF;
#pragma unroll
      for (int c = 0; c < 16; ++c) At[0][c][lane] = src[c];
      __builtin_prefetch(src + 32 * FF, 0, 1);       // global_prefetch_b8
    } else if (wave == 1) {
      const _Float16* src = sinB + krow * FF;
#pragma unroll
      for (int c = 0; c < 16; ++c) At[1][c][lane] = src[c];
      __builtin_prefetch(src + 32 * FF, 0, 1);
    }
    {
      const _Float16* src = hfB + krow * CC;
#pragma unroll
      for (int c = 0; c < 32; ++c) Bt[wave * 2 + (c >> 4)][c & 15][lane] = src[c];
      __builtin_prefetch(src + 32 * CC, 0, 1);
    }
    __syncthreads();
    v16h aC = gatherA_lds(At[0], lane);
    v16h aS = gatherA_lds(At[1], lane);
    v16h b0 = gatherB_lds(Bt[wave * 2 + 0], lane);
    v16h b1 = gatherB_lds(Bt[wave * 2 + 1], lane);
    aR0 = __builtin_amdgcn_wmma_f32_16x16x32_f16(false, aC, false, b0,
                                                 (short)0, aR0, false, false);
    aR1 = __builtin_amdgcn_wmma_f32_16x16x32_f16(false, aC, false, b1,
                                                 (short)0, aR1, false, false);
    aI0 = __builtin_amdgcn_wmma_f32_16x16x32_f16(false, aS, false, b0,
                                                 (short)0, aI0, false, false);
    aI1 = __builtin_amdgcn_wmma_f32_16x16x32_f16(false, aS, false, b1,
                                                 (short)0, aI1, false, false);
    __syncthreads();
  }
  // D layout: lane holds D[m = v + (lane>>4)*8][n' = lane&15]
  int nn = lane & 15;
  int mB = (lane >> 4) * 8;
  size_t base = (size_t)(ks * BB + b) * 2 * (FF * CC);
#pragma unroll
  for (int v = 0; v < 8; ++v) {
    size_t row = (size_t)(f0 + mB + v) * CC;
    size_t i0  = row + (c0 + nn);
    size_t i1  = row + (c0 + 16 + nn);
    hpart[base + i0]           = aR0[v];
    hpart[base + i1]           = aR1[v];
    hpart[base + FF * CC + i0] = aI0[v];
    hpart[base + FF * CC + i1] = aI1[v];
  }
}

// ---------------------------------------------------------------------------
// 4) deterministic split-K reduction + 1/N scale + phi_in sign (-i -> -sin)
// ---------------------------------------------------------------------------
__global__ void fno_reduceH(const float* __restrict__ hpart,
                            float* __restrict__ htr, float* __restrict__ hti) {
  int tid = blockIdx.x * blockDim.x + threadIdx.x;   // over B*F*C
  if (tid >= BB * FF * CC) return;
  int b = tid / (FF * CC);
  int j = tid - b * (FF * CC);
  float sr = 0.f, si = 0.f;
  for (int ks = 0; ks < KSPLIT; ++ks) {
    size_t base = (size_t)(ks * BB + b) * 2 * (FF * CC);
    sr += hpart[base + j];
    si += hpart[base + FF * CC + j];
  }
  const float invN = 1.0f / (float)NPTS;
  htr[tid] =  sr * invN;
  hti[tid] = -si * invN;                             // phi_in = exp(-i ang)
}

// ---------------------------------------------------------------------------
// 5) complex per-frequency channel mixing: G[b,o,f] = sum_c H[b,c,f]*W[c,o,f]
//    Stores Gt_r[f,o] and the NEGATED Gt_i[f,o] in f16 so the inverse pass is
//    two chained WMMAs into one accumulator: cos*Gr + sin*(-Gi).
// ---------------------------------------------------------------------------
__global__ void fno_mix(const float* __restrict__ htr, const float* __restrict__ hti,
                        const float* __restrict__ wr, const float* __restrict__ wi,
                        _Float16* __restrict__ gtr, _Float16* __restrict__ gtiN) {
  int tid = blockIdx.x * blockDim.x + threadIdx.x;   // over B*C*F (f fastest)
  if (tid >= BB * CC * FF) return;
  int f = tid & (FF - 1);
  int o = (tid >> 7) & (CC - 1);
  int b = tid >> 14;
  const float* hr = htr + (size_t)b * FF * CC + (size_t)f * CC;  // [c]
  const float* hi = hti + (size_t)b * FF * CC + (size_t)f * CC;
  float gr = 0.f, gi = 0.f;
  for (int c = 0; c < CC; ++c) {
    float wrv = wr[((size_t)c * CC + o) * FF + f];
    float wiv = wi[((size_t)c * CC + o) * FF + f];
    float hrv = hr[c], hiv = hi[c];
    gr += hrv * wrv - hiv * wiv;
    gi += hrv * wiv + hiv * wrv;
  }
  size_t oi = (size_t)b * FF * CC + (size_t)f * CC + o;          // Gt[f][o]
  gtr[oi]  = (_Float16)gr;
  gtiN[oi] = (_Float16)(-gi);
}

// ---------------------------------------------------------------------------
// 6) inverse evaluation + fused gelu:
//    hf[n,o] = gelu( sum_f cos[n,f]*Gr[f,o] + sin[n,f]*(-Gi[f,o]) )
//    M=n tile, N'=o tile, K=f (4 WMMA k-steps). Gt is invariant across
//    n-tiles: stage each per-wave k-slice through LDS ONCE, hoist the 8
//    gathered B fragments into registers, then sweep NTPB n-tiles with
//    direct-global A gathers + 8 chained WMMAs each. No barriers needed
//    (per-wave LDS buffers, same-wave DS ordering).
// ---------------------------------------------------------------------------
__global__ void __launch_bounds__(128) fno_inv(
    const _Float16* __restrict__ c16, const _Float16* __restrict__ s16,
    const _Float16* __restrict__ gtr, const _Float16* __restrict__ gtiN,
    float* __restrict__ hf32, _Float16* __restrict__ hf16, int storeHalf) {
  __shared__ _Float16 Br[4][16][34];
  __shared__ _Float16 Bi[4][16][34];
  int lane  = threadIdx.x & 31;
  int wave  = threadIdx.x >> 5;
  int nbase = blockIdx.x * (16 * NTPB);
  int o0    = (blockIdx.y * 4 + wave) * 16;
  int b     = blockIdx.z;
  const _Float16* grB = gtr  + (size_t)b * FF * CC + o0;
  const _Float16* giB = gtiN + (size_t)b * FF * CC + o0;

  // Stage Gt k-slices through per-wave LDS tile, hoist fragments to VGPRs.
  v16h bR[4], bI[4];
#pragma unroll
  for (int kstep = 0; kstep < 4; ++kstep) {
    int krow = kstep * 32 + lane;
    const _Float16* sr = grB + (size_t)krow * CC;
    const _Float16* si = giB + (size_t)krow * CC;
#pragma unroll
    for (int c = 0; c < 16; ++c) { Br[wave][c][lane] = sr[c]; Bi[wave][c][lane] = si[c]; }
    bR[kstep] = gatherB_lds(Br[wave], lane);
    bI[kstep] = gatherB_lds(Bi[wave], lane);
  }

  int m  = lane & 15;
  int kb = (lane >> 4) << 3;
  for (int nt = 0; nt < NTPB; ++nt) {
    int n0 = nbase + nt * 16;
    const _Float16* cr0 = c16 + (size_t)(b * NPTS + n0 + m) * FF;
    const _Float16* sw0 = s16 + (size_t)(b * NPTS + n0 + m) * FF;
    v8f acc = {};
#pragma unroll
    for (int kstep = 0; kstep < 4; ++kstep) {
      const _Float16* cr = cr0 + kstep * 32;
      const _Float16* sw = sw0 + kstep * 32;
      v16h aC, aS;
#pragma unroll
      for (int j = 0; j < 8; ++j) {
        int k = ((j >> 2) << 4) + kb + ((j & 3) << 1);
        aC[2 * j] = cr[k]; aC[2 * j + 1] = cr[k + 1];
        aS[2 * j] = sw[k]; aS[2 * j + 1] = sw[k + 1];
      }
      acc = __builtin_amdgcn_wmma_f32_16x16x32_f16(false, aC, false, bR[kstep],
                                                   (short)0, acc, false, false);
      acc = __builtin_amdgcn_wmma_f32_16x16x32_f16(false, aS, false, bI[kstep],
                                                   (short)0, acc, false, false);
    }
    int nn = lane & 15;
    int mB = (lane >> 4) * 8;
#pragma unroll
    for (int v = 0; v < 8; ++v) {
      float g = gelu_tanh(acc[v]);
      size_t idx = (size_t)(b * NPTS + n0 + mB + v) * CC + (o0 + nn);
      hf32[idx] = g;
      if (storeHalf) hf16[idx] = (_Float16)g;
    }
  }
}

// ---------------------------------------------------------------------------
// 7) fc_out: u = hf @ W_out + b   (C_OUT = 2)
// ---------------------------------------------------------------------------
__global__ void fno_fc_out(const float* __restrict__ hf32, const float* __restrict__ w,
                           const float* __restrict__ bias, float* __restrict__ out) {
  int row = blockIdx.x * blockDim.x + threadIdx.x;   // b*N + n
  if (row >= BB * NPTS) return;
  const float* h = hf32 + (size_t)row * CC;
  float a0 = bias[0], a1 = bias[1];
  for (int c = 0; c < CC; ++c) {
    float hv = h[c];
    a0 += hv * w[c * 2 + 0];
    a1 += hv * w[c * 2 + 1];
  }
  out[row * 2 + 0] = a0;
  out[row * 2 + 1] = a1;
}

// ---------------------------------------------------------------------------
extern "C" void kernel_launch(void* const* d_in, const int* in_sizes, int n_in,
                              void* d_out, int out_size, void* d_ws, size_t ws_size,
                              hipStream_t stream) {
  const float* a     = (const float*)d_in[0];
  const float* x     = (const float*)d_in[1];
  const float* y     = (const float*)d_in[2];
  const float* fiw   = (const float*)d_in[3];
  const float* fib   = (const float*)d_in[4];
  const float* freqs = (const float*)d_in[5];
  const float* wreal = (const float*)d_in[6];
  const float* wimag = (const float*)d_in[7];
  const float* fow   = (const float*)d_in[8];
  const float* fob   = (const float*)d_in[9];
  float* out = (float*)d_out;

  char* ws = (char*)d_ws;
  float*    hf32  = (float*)   (ws + OFF_HF32);
  _Float16* hf16  = (_Float16*)(ws + OFF_HF16);
  _Float16* cos16 = (_Float16*)(ws + OFF_COS);
  _Float16* sin16 = (_Float16*)(ws + OFF_SIN);
  float*    hpart = (float*)   (ws + OFF_HPART);
  float*    htr   = (float*)   (ws + OFF_HTR);
  float*    hti   = (float*)   (ws + OFF_HTI);
  _Float16* gtr   = (_Float16*)(ws + OFF_GTR);
  _Float16* gtiN  = (_Float16*)(ws + OFF_GTI);

  fno_fc_in<<<(BB * NPTS * CC + 255) / 256, 256, 0, stream>>>(a, fiw, fib, hf32, hf16);

  for (int l = 0; l < 4; ++l) {
    const float* kl = freqs + (size_t)l * FF * DD;
    const float* wr = wreal + (size_t)l * CC * CC * FF;
    const float* wi = wimag + (size_t)l * CC * CC * FF;

    fno_phi<<<(BB * NPTS * FF + 255) / 256, 256, 0, stream>>>(x, kl, cos16, sin16);
    fno_fwd<<<dim3(KSPLIT, 8, BB), 128, 0, stream>>>(hf16, cos16, sin16, hpart);
    fno_reduceH<<<(BB * FF * CC + 255) / 256, 256, 0, stream>>>(hpart, htr, hti);
    fno_mix<<<(BB * CC * FF + 255) / 256, 256, 0, stream>>>(htr, hti, wr, wi, gtr, gtiN);
    if (l == 3) {  // last conv queries at y: rebuild phase tables for y
      fno_phi<<<(BB * NPTS * FF + 255) / 256, 256, 0, stream>>>(y, kl, cos16, sin16);
    }
    fno_inv<<<dim3(NPTS / 16 / NTPB, 2, BB), 128, 0, stream>>>(cos16, sin16, gtr, gtiN,
                                                               hf32, hf16, (l < 3) ? 1 : 0);
  }

  fno_fc_out<<<(BB * NPTS + 255) / 256, 256, 0, stream>>>(hf32, fow, fob, out);
}
